// SpikingLinearAttention_18683107737651
// MI455X (gfx1250) — matrix operand order repaired and checked by reading, not compiled, not run on hardware
//
#include <hip/hip_runtime.h>
#include <hip/hip_bf16.h>
#include <math.h>

typedef __attribute__((ext_vector_type(16))) _Float16 v16h;
typedef __attribute__((ext_vector_type(8)))  _Float16 v8h;
typedef __attribute__((ext_vector_type(8)))  float    v8f;

#define LL 4096
#define BSZ 8
#define DD 512
#define HH 8
#define RR 48
#define DHD 64
#define NTOK (LL*BSZ)      // 32768
#define NQKV (3*DD)        // 1536

// ---------------- WMMA helpers (CDNA5 16x16x32 f16 -> f32) ----------------

__device__ __forceinline__ v8f wmma32(v16h a, v16h b, v8f c) {
  return __builtin_amdgcn_wmma_f32_16x16x32_f16(false, a, false, b, (short)0, c, false, false);
}

__device__ __forceinline__ v16h cat8(v8h lo, v8h hi) {
  return __builtin_shufflevector(lo, hi, 0,1,2,3,4,5,6,7,8,9,10,11,12,13,14,15);
}

// A fragment 16x32 f16 from row-major buffer (LDS or global).
// lane<16 -> row=lane, K=0..15; lane>=16 -> row=lane-16,
// K=16..31 (matches swizzled B). 16 contiguous halves -> two b128 loads.
// stride (in halves) must be a multiple of 8 and p 16B-aligned.
__device__ __forceinline__ v16h load_a_frag_v(const _Float16* p, int stride) {
  const int lane = threadIdx.x & 31;
  const _Float16* rp = p + (lane & 15) * stride + (lane >> 4) * 16;
  v8h lo = *(const v8h*)rp;
  v8h hi = *(const v8h*)(rp + 8);
  return cat8(lo, hi);
}

// B fragment from a fragment-swizzled 512-half block: lane's 16 halves are
// contiguous at block + lane*16 -> two b128 loads.
__device__ __forceinline__ v16h load_b_frag_sw(const _Float16* block) {
  const _Float16* rp = block + (threadIdx.x & 31) * 16;
  v8h lo = *(const v8h*)rp;
  v8h hi = *(const v8h*)(rp + 8);
  return cat8(lo, hi);
}

// fragment-swizzle index for element (k,n) of a [K][N] B matrix, ldn16 = N/16.
// block = (k/32, n/16); within block lane = ((k>>4)&1)*16 + (n&15), elem = k&15.
__device__ __forceinline__ size_t bswz(int k, int n, int ldn16) {
  return ((size_t)((k >> 5) * ldn16 + (n >> 4)) << 9)
       + ((size_t)((((k >> 4) & 1) << 4) + (n & 15)) << 4)
       + (size_t)(k & 15);
}

// ---------------- prep kernels ----------------

// Weights -> f16 in fragment-swizzled global layout (B = W^T, i.e. B[k][n]=W[n][k]).
__global__ void prep_weights(const float* __restrict__ Wq, const float* __restrict__ Wk,
                             const float* __restrict__ Wv, const float* __restrict__ Wo,
                             const float* __restrict__ Wrff,
                             _Float16* __restrict__ WqkvF, _Float16* __restrict__ WoF,
                             _Float16* __restrict__ WrffF) {
  int idx = blockIdx.x * 256 + threadIdx.x;
  if (idx < DD * NQKV) {
    int k = idx / NQKV, n = idx % NQKV;
    float w = (n < DD) ? Wq[n * DD + k]
            : (n < 2 * DD) ? Wk[(n - DD) * DD + k]
            : Wv[(n - 2 * DD) * DD + k];
    WqkvF[bswz(k, n, NQKV / 16)] = (_Float16)w;
  }
  if (idx < DD * DD) {
    int k = idx / DD, n = idx % DD;
    WoF[bswz(k, n, DD / 16)] = (_Float16)Wo[n * DD + k];
  }
  if (idx < HH * DHD * RR) {
    int h = idx / (DHD * RR), rem = idx % (DHD * RR);
    int k = rem / RR, n = rem % RR;     // Wrff[h] is already [K=dh][N=r]
    WrffF[(size_t)h * DHD * RR + bswz(k, n, RR / 16)] = (_Float16)Wrff[idx];
  }
}

__global__ void prep_params(const float* bnq_g, const float* bnq_b, const float* bnq_m, const float* bnq_v,
                            const float* bnk_g, const float* bnk_b, const float* bnk_m, const float* bnk_v,
                            const float* bnv_g, const float* bnv_b, const float* bnv_m, const float* bnv_v,
                            const float* Wv_b, const float* wtau_q, const float* wtau_k,
                            const float* brff_in,
                            float* __restrict__ scale, float* __restrict__ bias,
                            float* __restrict__ beta, float* __restrict__ brff_o) {
  int d = blockIdx.x * 256 + threadIdx.x;
  if (d < DD) {
    float sq = bnq_g[d] * rsqrtf(bnq_v[d] + 1e-5f);
    float sk = bnk_g[d] * rsqrtf(bnk_v[d] + 1e-5f);
    float sv = bnv_g[d] * rsqrtf(bnv_v[d] + 1e-5f);
    scale[d]          = sq;  bias[d]          = bnq_b[d] - bnq_m[d] * sq;
    scale[DD + d]     = sk;  bias[DD + d]     = bnk_b[d] - bnk_m[d] * sk;
    scale[2 * DD + d] = sv;  bias[2 * DD + d] = (Wv_b[d] - bnv_m[d]) * sv + bnv_b[d];
    float wq = wtau_q[d], wk = wtau_k[d];
    float spq = (wq > 20.f) ? wq : log1pf(__expf(wq));
    float spk = (wk > 20.f) ? wk : log1pf(__expf(wk));
    beta[d]      = fminf(fmaxf(1.f - 1.f / (spq + 1.f), 0.f), 0.9999f);
    beta[DD + d] = fminf(fmaxf(1.f - 1.f / (spk + 1.f), 0.f), 0.9999f);
  }
  if (d < HH * RR) brff_o[d] = brff_in[d];
}

// ---------------- generic 64x64 WMMA GEMM ----------------
// 128 thr = 4 waves; wave = 32x32 sub-tile (2x2 frags). A staged in LDS
// (vectorized b128 frag reads), B fragments loaded straight from swizzled
// global weights (two global b128 per frag, L2-resident).

template <bool SRC_F32>
__global__ void gemm64x64(const void* __restrict__ Aptr, const _Float16* __restrict__ Bsw,
                          int M, int N, int K,
                          const float* __restrict__ scale, const float* __restrict__ bias,
                          float* __restrict__ outF, _Float16* __restrict__ outH) {
  __shared__ __align__(16) _Float16 As[64][40];   // 40 halves = 80B rows (16B mult)
  const int tid = threadIdx.x;
  const int m0 = blockIdx.y * 64;
  const int n0 = blockIdx.x * 64;
  const int wid = tid >> 5;
  const int wm = (wid & 1) * 32, wn = (wid >> 1) * 32;
  const int ldn16 = N >> 4;

  v8f acc[2][2] = {};

  for (int k0 = 0; k0 < K; k0 += 32) {
    __syncthreads();
    {   // stage A 64x32 -> LDS, vector stores
      int r = tid >> 1, c0 = (tid & 1) * 16;
      if (SRC_F32) {
        const float* src = (const float*)Aptr + (size_t)(m0 + r) * K + k0 + c0;
        if (k0 + 32 < K) __builtin_prefetch(src + 32, 0, 1);
        v8h t0, t1;
#pragma unroll
        for (int i = 0; i < 8; ++i) { t0[i] = (_Float16)src[i]; t1[i] = (_Float16)src[8 + i]; }
        *(v8h*)&As[r][c0] = t0;
        *(v8h*)&As[r][c0 + 8] = t1;
      } else {
        const _Float16* src = (const _Float16*)Aptr + (size_t)(m0 + r) * K + k0 + c0;
        if (k0 + 32 < K) __builtin_prefetch(src + 32, 0, 1);
        *(v8h*)&As[r][c0]     = *(const v8h*)src;
        *(v8h*)&As[r][c0 + 8] = *(const v8h*)(src + 8);
      }
    }
    __syncthreads();
    const int kt = k0 >> 5;
    v16h a0 = load_a_frag_v(&As[wm][0], 40);
    v16h a1 = load_a_frag_v(&As[wm + 16][0], 40);
#pragma unroll
    for (int ni = 0; ni < 2; ++ni) {
      int nt = ((n0 + wn) >> 4) + ni;
      v16h b = load_b_frag_sw(Bsw + (((size_t)kt * ldn16 + nt) << 9));
      acc[0][ni] = wmma32(a0, b, acc[0][ni]);
      acc[1][ni] = wmma32(a1, b, acc[1][ni]);
    }
  }

  const int lane = tid & 31, half = lane >> 4, col = lane & 15;
#pragma unroll
  for (int mi = 0; mi < 2; ++mi)
#pragma unroll
    for (int ni = 0; ni < 2; ++ni)
#pragma unroll
      for (int j = 0; j < 8; ++j) {
        int m = m0 + wm + mi * 16 + half * 8 + j;
        int n = n0 + wn + ni * 16 + col;
        float v = acc[mi][ni][j];
        if (scale) v = v * scale[n] + bias[n];
        else if (bias) v += bias[n];
        if (outH) outH[(size_t)m * N + n] = (_Float16)v;
        else      outF[(size_t)m * N + n] = v;
      }
}

// ---------------- membrane EMA (in-place on f16 qkv buffer) ----------------
__global__ void membrane_kernel(_Float16* __restrict__ qkv, const float* __restrict__ beta) {
  int ch = blockIdx.x * 256 + threadIdx.x;          // [0,4096)
  int sel = blockIdx.y;                             // 0 = q, 1 = k
  int b = ch >> 9, d = ch & 511;
  float be = beta[sel * DD + d];
  float om = 1.f - be;
  float m = 0.f;
  size_t idx = (size_t)b * NQKV + sel * DD + d;
  const size_t stride = (size_t)BSZ * NQKV;
  for (int l = 0; l < LL; ++l) {
    float x = (float)qkv[idx];
    m = be * m + om * x;
    qkv[idx] = (_Float16)m;
    idx += stride;
  }
}

// ---------------- RFF phi ----------------
// per (b,h,l-chunk): [64 l x 64 k] x [64 k x 48 r]. phi_q written l-major
// [bh][l][48]; phi_k written TRANSPOSED [bh][r][L] (contiguous v8h stores,
// and makes the kv reduction's A-fragments contiguous global loads).
__global__ void rff_kernel(const _Float16* __restrict__ qkv, const _Float16* __restrict__ WrffF,
                           const float* __restrict__ brff,
                           _Float16* __restrict__ phi_q, _Float16* __restrict__ phiT_k) {
  __shared__ __align__(16) _Float16 Qs[64][72];
  const int tid = threadIdx.x;
  const int l0 = blockIdx.x * 64;
  const int bh = blockIdx.y;
  const int b = bh >> 3, h = bh & 7;
  const int sel = blockIdx.z;
  const int colOff = sel * DD + h * DHD;
  {
    int r = tid >> 1, c0 = (tid & 1) * 32;
    const _Float16* src = qkv + ((size_t)(l0 + r) * BSZ + b) * NQKV + colOff + c0;
#pragma unroll
    for (int i = 0; i < 4; ++i)
      *(v8h*)&Qs[r][c0 + i * 8] = *(const v8h*)(src + i * 8);
  }
  __syncthreads();
  const int w = tid >> 5;
  const _Float16* wb = WrffF + (size_t)h * DHD * RR;
  v8f acc[3] = {};
#pragma unroll
  for (int kt = 0; kt < 2; ++kt) {
    v16h a = load_a_frag_v(&Qs[w * 16][kt * 32], 72);
#pragma unroll
    for (int ni = 0; ni < 3; ++ni) {
      v16h bf = load_b_frag_sw(wb + (((size_t)kt * 3 + ni) << 9));
      acc[ni] = wmma32(a, bf, acc[ni]);
    }
  }
  const int lane = tid & 31, half = lane >> 4, col = lane & 15;
  const float inv = 0.14433756729740643f;           // 1/sqrt(48)
#pragma unroll
  for (int ni = 0; ni < 3; ++ni) {
    int r = ni * 16 + col;
    int lbase = l0 + w * 16 + half * 8;
    float bconst = brff[h * RR + r];
    v8h ph;
#pragma unroll
    for (int j = 0; j < 8; ++j) {
      float p = acc[ni][j] + bconst;
      float e = (p > 0.f) ? p : (__expf(p) - 1.f);  // elu
      ph[j] = (_Float16)((e + 1.f + 1e-6f) * inv);
    }
    if (sel == 0) {
#pragma unroll
      for (int j = 0; j < 8; ++j)
        phi_q[((size_t)bh * LL + lbase + j) * RR + r] = ph[j];
    } else {
      *(v8h*)(phiT_k + ((size_t)bh * RR + r) * LL + lbase) = ph;  // contiguous
    }
  }
}

// ---------------- KV_sum / K_sum ----------------
// per (b,h): KV[r][d] = sum_l phiT_k[r][l] * v[l][d]; K = 4096 in 32-chunks.
// A fragments: contiguous b128 loads straight from global phiT_k.
// B (v chunk): staged into LDS pre-swizzled to fragment layout.
__global__ void kv_kernel(const _Float16* __restrict__ phiT_k, const _Float16* __restrict__ qkv,
                          float* __restrict__ KV, float* __restrict__ Ksum) {
  __shared__ __align__(16) _Float16 VsF[4 * 512];   // 4 n-tiles x (32 lanes x 16)
  const int tid = threadIdx.x;
  const int bh = blockIdx.x;
  const int b = bh >> 3, h = bh & 7;
  const int w = tid >> 5, lane = tid & 31, half = lane >> 4, col = lane & 15;
  v8f acc[3] = {};
  for (int l0 = 0; l0 < LL; l0 += 32) {
    __syncthreads();
    {   // stage v[l0..l0+32)[h*64..+64) swizzled: element (k=l-l0, d)
      int r = tid >> 2, c0 = (tid & 3) * 16;
      const _Float16* src = qkv + ((size_t)(l0 + r) * BSZ + b) * NQKV + 2 * DD + h * DHD + c0;
      if (l0 + 32 < LL) __builtin_prefetch(src + (size_t)32 * BSZ * NQKV, 0, 1);
      v8h t0 = *(const v8h*)src;
      v8h t1 = *(const v8h*)(src + 8);
      int lanebase = ((r >> 4) << 4), elem = r & 15;
#pragma unroll
      for (int i = 0; i < 8; ++i) {
        int d0 = c0 + i, d1 = c0 + 8 + i;
        VsF[((d0 >> 4) << 9) + ((lanebase + (d0 & 15)) << 4) + elem] = t0[i];
        VsF[((d1 >> 4) << 9) + ((lanebase + (d1 & 15)) << 4) + elem] = t1[i];
      }
    }
    __syncthreads();
    v16h bf = load_b_frag_sw(&VsF[w << 9]);
#pragma unroll
    for (int mi = 0; mi < 3; ++mi) {
      const _Float16* ap = phiT_k + ((size_t)bh * RR + mi * 16 + (lane & 15)) * LL
                         + l0 + (lane >> 4) * 16;
      v16h a = cat8(*(const v8h*)ap, *(const v8h*)(ap + 8));
      acc[mi] = wmma32(a, bf, acc[mi]);
    }
  }
#pragma unroll
  for (int mi = 0; mi < 3; ++mi)
#pragma unroll
    for (int j = 0; j < 8; ++j) {
      int r = mi * 16 + half * 8 + j;
      int d = w * 16 + col;
      KV[((size_t)bh * RR + r) * DHD + d] = acc[mi][j];
    }
  if (tid < RR) {   // K_sum: row sums of phiT_k (contiguous reads)
    float ks = 0.f;
    const _Float16* rp = phiT_k + ((size_t)bh * RR + tid) * LL;
    for (int l = 0; l < LL; ++l) ks += (float)rp[l];
    Ksum[bh * RR + tid] = fmaxf(ks, 1e-6f);
  }
}

// ---------------- num/den + merge heads ----------------
__global__ void attn_kernel(const _Float16* __restrict__ phi_q, const float* __restrict__ KV,
                            const float* __restrict__ Ksum, _Float16* __restrict__ attn) {
  __shared__ __align__(16) _Float16 As[64][72];     // phi_q rows, K padded 48->64
  __shared__ __align__(16) _Float16 KVsF[8 * 512];  // 2 k-tiles x 4 n-tiles, swizzled
  __shared__ float dens[64];
  const int tid = threadIdx.x;
  const int l0 = blockIdx.x * 64;
  const int bh = blockIdx.y;
  const int b = bh >> 3, h = bh & 7;
  {   // stage phi_q rows (48 valid halves, pad to 64)
    int r = tid >> 1, c0 = (tid & 1) * 32;
    const _Float16* src = phi_q + ((size_t)bh * LL + l0 + r) * RR;
    if (c0 == 0) {
#pragma unroll
      for (int i = 0; i < 4; ++i)
        *(v8h*)&As[r][i * 8] = *(const v8h*)(src + i * 8);
    } else {
      *(v8h*)&As[r][32] = *(const v8h*)(src + 32);
      *(v8h*)&As[r][40] = *(const v8h*)(src + 40);
      v8h z = {};
      *(v8h*)&As[r][48] = z;
      *(v8h*)&As[r][56] = z;
    }
  }
  {   // stage KV (48x64 fp32 -> f16) swizzled, zero pad k=48..63
    for (int e = tid; e < 64 * 64; e += 128) {
      int k = e >> 6, d = e & 63;
      float v = (k < RR) ? KV[((size_t)bh * RR + k) * DHD + d] : 0.f;
      KVsF[(((k >> 5) << 2) + (d >> 4)) * 512
           + (((((k >> 4) & 1) << 4) + (d & 15)) << 4) + (k & 15)] = (_Float16)v;
    }
  }
  __syncthreads();
  if (tid < 64) {
    float dv = 1e-6f;
    const float* kp = Ksum + bh * RR;
#pragma unroll
    for (int r = 0; r < RR; ++r) dv += (float)As[tid][r] * kp[r];
    dens[tid] = dv;
  }
  __syncthreads();
  const int w = tid >> 5, lane = tid & 31, half = lane >> 4, col = lane & 15;
  v8f acc[4] = {};
#pragma unroll
  for (int kt = 0; kt < 2; ++kt) {
    v16h a = load_a_frag_v(&As[w * 16][kt * 32], 72);
#pragma unroll
    for (int ni = 0; ni < 4; ++ni) {
      v16h bf = load_b_frag_sw(&KVsF[((kt << 2) + ni) << 9]);
      acc[ni] = wmma32(a, bf, acc[ni]);
    }
  }
#pragma unroll
  for (int ni = 0; ni < 4; ++ni)
#pragma unroll
    for (int j = 0; j < 8; ++j) {
      int lr = w * 16 + half * 8 + j;
      int d = ni * 16 + col;
      float v = acc[ni][j] / dens[lr];
      attn[((size_t)(l0 + lr) * BSZ + b) * DD + h * DHD + d] = (_Float16)v;
    }
}

// ---------------- host launch ----------------

extern "C" void kernel_launch(void* const* d_in, const int* in_sizes, int n_in,
                              void* d_out, int out_size, void* d_ws, size_t ws_size,
                              hipStream_t stream) {
  (void)in_sizes; (void)n_in; (void)out_size; (void)ws_size;
  const float* x      = (const float*)d_in[0];
  const float* Wq     = (const float*)d_in[1];
  const float* Wk     = (const float*)d_in[2];
  const float* Wv     = (const float*)d_in[3];
  const float* Wv_b   = (const float*)d_in[4];
  const float* Wo     = (const float*)d_in[5];
  const float* Wo_b   = (const float*)d_in[6];
  const float* bnq_g  = (const float*)d_in[7];
  const float* bnq_b  = (const float*)d_in[8];
  const float* bnq_m  = (const float*)d_in[9];
  const float* bnq_v  = (const float*)d_in[10];
  const float* bnk_g  = (const float*)d_in[11];
  const float* bnk_b  = (const float*)d_in[12];
  const float* bnk_m  = (const float*)d_in[13];
  const float* bnk_v  = (const float*)d_in[14];
  const float* bnv_g  = (const float*)d_in[15];
  const float* bnv_b  = (const float*)d_in[16];
  const float* bnv_m  = (const float*)d_in[17];
  const float* bnv_v  = (const float*)d_in[18];
  const float* wtau_q = (const float*)d_in[19];
  const float* wtau_k = (const float*)d_in[20];
  const float* Wrff   = (const float*)d_in[21];
  const float* brff   = (const float*)d_in[22];

  char* p = (char*)d_ws;
  auto alloc = [&](size_t bytes) {
    char* r = p;
    p += (bytes + 255) & ~(size_t)255;
    return r;
  };
  _Float16* WqkvF  = (_Float16*)alloc((size_t)DD * NQKV * 2);
  _Float16* WoF    = (_Float16*)alloc((size_t)DD * DD * 2);
  _Float16* WrffF  = (_Float16*)alloc((size_t)HH * DHD * RR * 2);
  float* scale     = (float*)alloc((size_t)NQKV * 4);
  float* bias      = (float*)alloc((size_t)NQKV * 4);
  float* beta      = (float*)alloc((size_t)2 * DD * 4);
  float* brff_o    = (float*)alloc((size_t)HH * RR * 4);
  _Float16* qkv    = (_Float16*)alloc((size_t)NTOK * NQKV * 2);        // 96 MB
  _Float16* phi_q  = (_Float16*)alloc((size_t)BSZ * HH * LL * RR * 2); // 25 MB
  _Float16* phiT_k = (_Float16*)alloc((size_t)BSZ * HH * RR * LL * 2); // 25 MB
  float* KV        = (float*)alloc((size_t)BSZ * HH * RR * DHD * 4);
  float* Ksum      = (float*)alloc((size_t)BSZ * HH * RR * 4);
  _Float16* attn   = (_Float16*)alloc((size_t)NTOK * DD * 2);          // 32 MB

  prep_weights<<<(DD * NQKV + 255) / 256, 256, 0, stream>>>(Wq, Wk, Wv, Wo, Wrff,
                                                            WqkvF, WoF, WrffF);
  prep_params<<<2, 256, 0, stream>>>(bnq_g, bnq_b, bnq_m, bnq_v,
                                     bnk_g, bnk_b, bnk_m, bnk_v,
                                     bnv_g, bnv_b, bnv_m, bnv_v,
                                     Wv_b, wtau_q, wtau_k, brff,
                                     scale, bias, beta, brff_o);
  gemm64x64<true><<<dim3(NQKV / 64, NTOK / 64), 128, 0, stream>>>(
      (const void*)x, WqkvF, NTOK, NQKV, DD, scale, bias, nullptr, qkv);
  membrane_kernel<<<dim3(BSZ * DD / 256, 2), 256, 0, stream>>>(qkv, beta);
  rff_kernel<<<dim3(LL / 64, BSZ * HH, 2), 128, 0, stream>>>(qkv, WrffF, brff_o, phi_q, phiT_k);
  kv_kernel<<<BSZ * HH, 128, 0, stream>>>(phiT_k, qkv, KV, Ksum);
  attn_kernel<<<dim3(LL / 64, BSZ * HH), 128, 0, stream>>>(phi_q, KV, Ksum, attn);
  gemm64x64<false><<<dim3(DD / 64, NTOK / 64), 128, 0, stream>>>(
      (const void*)attn, WoF, NTOK, DD, DD, nullptr, Wo_b, (float*)d_out, nullptr);
}
